// SPCausalSelfAttention_34067680592426
// MI455X (gfx1250) — compile-verified, matrix operand order checked by reading, not run on hardware
//
#include <hip/hip_runtime.h>
#include <hip/hip_bf16.h>

// ---------------------------------------------------------------------------
// Causal self-attention block for MI455X (gfx1250, wave32, WMMA + TDM).
//   B=2, S=2048, E=1024, H=16, D=64
// Pipeline:
//   0) one-shot fp32 -> bf16 conversion of x, W_attn, W_proj (bandwidth-bound)
//   1) qkv(bf16) = xb @ Wab + b_attn      [WMMA GEMM, TDM-staged LDS tiles]
//   2) y(bf16)   = flash-attention(qkv)   [WMMA, global b128 frag loads]
//   3) out(f32)  = y @ Wpb + b_proj       [WMMA GEMM, TDM-staged LDS tiles]
// ---------------------------------------------------------------------------

typedef __attribute__((ext_vector_type(16))) __bf16 v16bf;
typedef __attribute__((ext_vector_type(8)))  float  v8f;
typedef unsigned int u32x4 __attribute__((ext_vector_type(4)));
typedef int          i32x8 __attribute__((ext_vector_type(8)));
typedef int          i32x4 __attribute__((ext_vector_type(4)));

#define BDIM 2
#define SDIM 2048
#define EDIM 1024
#define HDIM 16
#define DDIM 64
#define RS   (3 * EDIM)
#define MROWS (BDIM * SDIM)

#if defined(__AMDGCN__) && __has_builtin(__builtin_amdgcn_tensor_load_to_lds)
#define USE_TDM 1
#else
#define USE_TDM 0
#endif

#if USE_TDM
// Issue a 2D-tile TDM load: global (row-major, bf16) -> LDS with row padding.
// tile_w elems per row (contiguous), tile_h rows; LDS rows padded per the D#
// pad feature so WMMA fragment reads are bank-conflict-free and 16B aligned.
__device__ __forceinline__ void tdm_load_tile_2d(
    unsigned lds_off, const __bf16* gptr, unsigned tensor_w, unsigned tensor_h,
    unsigned stride0_elems, unsigned tile_w, unsigned tile_h,
    unsigned pad_interval_code, unsigned pad_amount_code) {
  unsigned long long ga = (unsigned long long)gptr;
  u32x4 g0;
  g0.x = 1u;                                           // count=1, user mode
  g0.y = lds_off;                                      // lds_addr
  g0.z = (unsigned)(ga & 0xffffffffu);                 // global_addr[31:0]
  g0.w = (unsigned)((ga >> 32) & 0x01ffffffu) | (2u << 30);  // addr[56:32]|type=2
  i32x8 g1;
  g1[0] = (int)((1u << 16)                             // data_size = 2 bytes
                | (1u << 20)                           // pad_enable
                | (pad_interval_code << 22)
                | (pad_amount_code << 25));
  g1[1] = (int)((tensor_w & 0xffffu) << 16);           // tensor_dim0[15:0]
  g1[2] = (int)((tensor_w >> 16) | ((tensor_h & 0xffffu) << 16));
  g1[3] = (int)(((tensor_h >> 16) & 0xffffu) | (tile_w << 16));  // tile_dim0
  g1[4] = (int)tile_h;                                 // tile_dim1 (tile_dim2=0)
  g1[5] = (int)stride0_elems;                          // tensor_dim0_stride lo32
  g1[6] = 0;
  g1[7] = 0;
  i32x4 z4 = {};
#if defined(__clang_major__) && (__clang_major__ >= 23)
  i32x8 z8 = {};
  __builtin_amdgcn_tensor_load_to_lds(g0, g1, z4, z4, z8, 0);
#else
  __builtin_amdgcn_tensor_load_to_lds(g0, g1, z4, z4, 0);
#endif
}
#endif  // USE_TDM

// ---------------------------------------------------------------------------
// One-shot fp32 -> bf16 conversion (vectorized, bandwidth-bound).
// ---------------------------------------------------------------------------
__global__ __launch_bounds__(256) void cvt_f32_bf16(
    const float* __restrict__ src, __bf16* __restrict__ dst, int n4) {
  int i = blockIdx.x * 256 + threadIdx.x;
  if (i < n4) {
    float4 f = ((const float4*)src)[i];
    union { __bf16 b[4]; uint2 u; } o;
    o.b[0] = (__bf16)f.x; o.b[1] = (__bf16)f.y;
    o.b[2] = (__bf16)f.z; o.b[3] = (__bf16)f.w;
    ((uint2*)dst)[i] = o.u;
  }
}

// ---------------------------------------------------------------------------
// Tiled bf16 GEMM: C[M,N] = A[M,K] @ Bm[K,N] + bias[N]
// Block tile 128x128, BK=32, 256 threads = 8 waves (4 along M x 2 along N),
// each wave computes 32x64 = 2x4 WMMA frags: 8 WMMAs per 6 fragment reads.
// Tiles staged to LDS by the Tensor Data Mover (wave 0), padded rows.
// ---------------------------------------------------------------------------
template <typename OT>
__global__ __launch_bounds__(256) void gemm128_wmma_bf16(
    const __bf16* __restrict__ A, const __bf16* __restrict__ Bm,
    const float* __restrict__ bias, OT* __restrict__ C,
    int M, int N, int K) {
  // Row pads: 32 -> 40 elems (pad 4 dwords per 16 dwords), 128 -> 136.
  __shared__ __align__(16) __bf16 As[128][40];
  __shared__ __align__(16) __bf16 Bs[32][136];

  const int tid  = threadIdx.x;
  const int wave = tid >> 5;
  const int lane = tid & 31;
  const int m16  = lane & 15;
  const int hi8  = (lane >= 16) ? 8 : 0;
  const int bm   = blockIdx.y * 128;
  const int bn   = blockIdx.x * 128;
  const int wm   = (wave & 3) * 32;   // wave row offset
  const int wn   = (wave >> 2) * 64;  // wave col offset

  v8f acc[2][4] = {};

  for (int k0 = 0; k0 < K; k0 += 32) {
#if USE_TDM
    if (tid < 32) {
      // A tile: 128 rows x 32 elems; pad_interval 16 dwords (code 3),
      // pad_amount 4 dwords (code 3) -> LDS row stride 40 elems.
      tdm_load_tile_2d((unsigned)(size_t)&As[0][0],
                       A + (size_t)bm * K + k0, (unsigned)K, (unsigned)M,
                       (unsigned)K, 32u, 128u, 3u, 3u);
      // B tile: 32 rows x 128 elems; pad_interval 64 dwords (code 5),
      // pad_amount 4 dwords (code 3) -> LDS row stride 136 elems.
      tdm_load_tile_2d((unsigned)(size_t)&Bs[0][0],
                       Bm + (size_t)k0 * N + bn, (unsigned)N, (unsigned)K,
                       (unsigned)N, 128u, 32u, 5u, 3u);
      __builtin_amdgcn_s_wait_tensorcnt(0);
    }
    __syncthreads();
#else
    // Fallback staging: 16B vector copies, 256 threads, 2 chunks each side.
#pragma unroll
    for (int j = 0; j < 2; ++j) {
      int chunk = tid + 256 * j;            // 512 chunks of 8 bf16
      int r = chunk >> 2, c = (chunk & 3) * 8;
      *(uint4*)&As[r][c] =
          *(const uint4*)(A + (size_t)(bm + r) * K + k0 + c);
      int r2 = chunk >> 4, c2 = (chunk & 15) * 8;
      *(uint4*)&Bs[r2][c2] =
          *(const uint4*)(Bm + (size_t)(k0 + r2) * N + bn + c2);
    }
    __syncthreads();
#endif

    // Fragments per ISA 16-bit layout: lane halves hold K={0..7,16..23} /
    // {8..15,24..31}. A-frag reads are two contiguous 16B chunks per lane.
    v16bf af[2], bfrag[4];
#pragma unroll
    for (int i = 0; i < 2; ++i) {
      int row = wm + i * 16 + m16;
#pragma unroll
      for (int j = 0; j < 8; ++j) {
        af[i][j]     = As[row][hi8 + j];
        af[i][j + 8] = As[row][16 + hi8 + j];
      }
    }
#pragma unroll
    for (int jj = 0; jj < 4; ++jj) {
      int col = wn + jj * 16 + m16;
#pragma unroll
      for (int j = 0; j < 8; ++j) {
        bfrag[jj][j]     = Bs[hi8 + j][col];
        bfrag[jj][j + 8] = Bs[16 + hi8 + j][col];
      }
    }
#pragma unroll
    for (int i = 0; i < 2; ++i)
#pragma unroll
      for (int jj = 0; jj < 4; ++jj)
        acc[i][jj] = __builtin_amdgcn_wmma_f32_16x16x32_bf16(
            false, af[i], false, bfrag[jj], (short)0, acc[i][jj], false, false);
    __syncthreads();
  }

  // Epilogue: C-frag -> lane holds col N=lane%16, rows e+hi8.
#pragma unroll
  for (int jj = 0; jj < 4; ++jj) {
    int col = bn + wn + jj * 16 + m16;
    float bv = bias[col];
#pragma unroll
    for (int i = 0; i < 2; ++i)
#pragma unroll
      for (int e = 0; e < 8; ++e) {
        int row = bm + wm + i * 16 + e + hi8;
        C[(size_t)row * N + col] = (OT)(acc[i][jj][e] + bv);
      }
  }
}

// ---------------------------------------------------------------------------
// Flash attention: one wave per (b, h, 16-query tile). Online softmax.
// ---------------------------------------------------------------------------
__global__ __launch_bounds__(32) void attn_flash_wmma(
    const __bf16* __restrict__ qkv, __bf16* __restrict__ y) {
  const int qt   = blockIdx.x;
  const int h    = blockIdx.y;
  const int b    = blockIdx.z;
  const int lane = threadIdx.x;
  const int q0   = qt * 16;
  const int m16  = lane & 15;
  const int hi8  = (lane >= 16) ? 8 : 0;

  __shared__ __align__(16) __bf16 Pt[16][36];

  v16bf qf[2];
  {
    const __bf16* qrow = qkv + ((size_t)b * SDIM + q0 + m16) * RS + h * DDIM;
#pragma unroll
    for (int f = 0; f < 2; ++f)
#pragma unroll
      for (int j = 0; j < 8; ++j) {
        qf[f][j]     = qrow[f * 32 + hi8 + j];
        qf[f][j + 8] = qrow[f * 32 + 16 + hi8 + j];
      }
  }

  float mrow[8], lrow[8];
  v8f acc[4] = {};
#pragma unroll
  for (int e = 0; e < 8; ++e) { mrow[e] = -3.0e38f; lrow[e] = 0.0f; }
  const float scale = 0.125f;  // 1/sqrt(64)

  for (int kblk = 0; kblk < q0 + 16; kblk += 32) {
    v8f sfr[2];
#pragma unroll
    for (int c = 0; c < 2; ++c) {
      const __bf16* kptr =
          qkv + ((size_t)b * SDIM + kblk + c * 16 + m16) * RS + EDIM + h * DDIM;
      v16bf bk0, bk1;
#pragma unroll
      for (int j = 0; j < 8; ++j) {
        bk0[j]     = kptr[hi8 + j];
        bk0[j + 8] = kptr[16 + hi8 + j];
        bk1[j]     = kptr[32 + hi8 + j];
        bk1[j + 8] = kptr[48 + hi8 + j];
      }
      v8f s = {};
      s = __builtin_amdgcn_wmma_f32_16x16x32_bf16(false, qf[0], false, bk0,
                                                  (short)0, s, false, false);
      s = __builtin_amdgcn_wmma_f32_16x16x32_bf16(false, qf[1], false, bk1,
                                                  (short)0, s, false, false);
      sfr[c] = s;
    }

    float p0[8], p1[8];
#pragma unroll
    for (int e = 0; e < 8; ++e) {
      int qg = q0 + e + hi8;
      p0[e] = (kblk + m16      <= qg) ? sfr[0][e] * scale : -3.0e38f;
      p1[e] = (kblk + 16 + m16 <= qg) ? sfr[1][e] * scale : -3.0e38f;
    }
#pragma unroll
    for (int e = 0; e < 8; ++e) {
      float v = fmaxf(p0[e], p1[e]);
#pragma unroll
      for (int off = 1; off < 16; off <<= 1)
        v = fmaxf(v, __shfl_xor(v, off, 32));
      float mnew  = fmaxf(mrow[e], v);
      float alpha = __expf(mrow[e] - mnew);
      mrow[e] = mnew;
      p0[e] = __expf(p0[e] - mnew);
      p1[e] = __expf(p1[e] - mnew);
      float rs = p0[e] + p1[e];
#pragma unroll
      for (int off = 1; off < 16; off <<= 1)
        rs += __shfl_xor(rs, off, 32);
      lrow[e] = lrow[e] * alpha + rs;
#pragma unroll
      for (int d = 0; d < 4; ++d) acc[d][e] *= alpha;
    }

#pragma unroll
    for (int e = 0; e < 8; ++e) {
      Pt[e + hi8][m16]      = (__bf16)p0[e];
      Pt[e + hi8][16 + m16] = (__bf16)p1[e];
    }
    __syncthreads();
    v16bf pf;
#pragma unroll
    for (int j = 0; j < 8; ++j) {
      pf[j]     = Pt[m16][hi8 + j];
      pf[j + 8] = Pt[m16][16 + hi8 + j];
    }
    __syncthreads();

#pragma unroll
    for (int d = 0; d < 4; ++d) {
      const __bf16* vbase = qkv + ((size_t)b * SDIM + kblk) * RS + 2 * EDIM +
                            h * DDIM + d * 16 + m16;
      v16bf vf;
#pragma unroll
      for (int j = 0; j < 8; ++j) {
        vf[j]     = vbase[(size_t)(hi8 + j) * RS];
        vf[j + 8] = vbase[(size_t)(16 + hi8 + j) * RS];
      }
      acc[d] = __builtin_amdgcn_wmma_f32_16x16x32_bf16(
          false, pf, false, vf, (short)0, acc[d], false, false);
    }
  }

#pragma unroll
  for (int d = 0; d < 4; ++d)
#pragma unroll
    for (int e = 0; e < 8; ++e) {
      int row = q0 + e + hi8;
      float v = acc[d][e] / lrow[e];
      y[((size_t)b * SDIM + row) * EDIM + h * DDIM + d * 16 + m16] = (__bf16)v;
    }
}

// ---------------------------------------------------------------------------
extern "C" void kernel_launch(void* const* d_in, const int* in_sizes, int n_in,
                              void* d_out, int out_size, void* d_ws,
                              size_t ws_size, hipStream_t stream) {
  (void)in_sizes; (void)n_in; (void)out_size; (void)ws_size;
  const float* x  = (const float*)d_in[0];
  const float* Wa = (const float*)d_in[1];
  const float* ba = (const float*)d_in[2];
  const float* Wp = (const float*)d_in[3];
  const float* bp = (const float*)d_in[4];
  float* out = (float*)d_out;

  // Workspace layout (bf16): qkv | y | xb | Wab | Wpb  (~48 MB total)
  __bf16* qkv = (__bf16*)d_ws;
  __bf16* y   = qkv + (size_t)MROWS * RS;
  __bf16* xb  = y   + (size_t)MROWS * EDIM;
  __bf16* Wab = xb  + (size_t)MROWS * EDIM;
  __bf16* Wpb = Wab + (size_t)EDIM * 3 * EDIM;

  // 0) one-shot fp32 -> bf16 conversions
  int nx = MROWS * EDIM / 4, na = EDIM * 3 * EDIM / 4, np = EDIM * EDIM / 4;
  cvt_f32_bf16<<<(nx + 255) / 256, 256, 0, stream>>>(x, xb, nx);
  cvt_f32_bf16<<<(na + 255) / 256, 256, 0, stream>>>(Wa, Wab, na);
  cvt_f32_bf16<<<(np + 255) / 256, 256, 0, stream>>>(Wp, Wpb, np);

  // 1) QKV GEMM: (4096,1024) @ (1024,3072) + b_attn -> bf16 qkv
  dim3 g1(3 * EDIM / 128, MROWS / 128);
  gemm128_wmma_bf16<__bf16><<<g1, 256, 0, stream>>>(
      xb, Wab, ba, qkv, MROWS, 3 * EDIM, EDIM);

  // 2) Flash attention per (b, h, 16-row q tile)
  dim3 g2(SDIM / 16, HDIM, BDIM);
  attn_flash_wmma<<<g2, 32, 0, stream>>>(qkv, y);

  // 3) Projection GEMM: (4096,1024) @ (1024,1024) + b_proj -> f32 out
  dim3 g3(EDIM / 128, MROWS / 128);
  gemm128_wmma_bf16<float><<<g3, 256, 0, stream>>>(
      y, Wpb, bp, out, MROWS, EDIM, EDIM);
}
